// zhannRNN_71262097375385
// MI455X (gfx1250) — matrix operand order repaired and checked by reading, not compiled, not run on hardware
//
#include <hip/hip_runtime.h>
#include <hip/hip_bf16.h>
#include <math.h>

typedef __attribute__((ext_vector_type(16))) _Float16 v16h;
typedef __attribute__((ext_vector_type(8)))  _Float16 v8h;
typedef __attribute__((ext_vector_type(8)))  float    v8f;
typedef __attribute__((ext_vector_type(4)))  unsigned v4u;
typedef __attribute__((ext_vector_type(8)))  int      v8i;
typedef __attribute__((ext_vector_type(4)))  int      v4i;

#define IN_DIM  512
#define HID     2048
#define OUT_DIM 128
#define BATCH   8
#define TSTEPS  512
#define MROWS   (BATCH * TSTEPS)   // 4096

// ---------------- helpers ----------------

static __device__ inline v16h cat8(v8h lo, v8h hi) {
  return __builtin_shufflevector(lo, hi, 0,1,2,3,4,5,6,7,8,9,10,11,12,13,14,15);
}

// A matrix (16x32 f16) from row-major src (16 rows, 'stride' halves apart).
// Lane L<16: row=L, K=0..15; lane L>=16: row=L-16, K=16..31 (contiguous).
static __device__ inline v16h load_A_f16(const _Float16* base, int stride, int lane, int k0) {
  const int row = lane & 15;
  const int hi  = lane >> 4;
  const v8h* p0 = (const v8h*)(base + (size_t)row * stride + k0 + hi * 16);
  const v8h* p1 = (const v8h*)(base + (size_t)row * stride + k0 + hi * 16 + 8);
  return cat8(*p0, *p1);
}

// B matrix (32x16 f16), B[k][n] = W[n][k], W row-major with 'stride' halves.
// Lane L<16: col=L, K=k0..k0+15; lane L>=16: col=L-16, K=k0+16..k0+31 (contiguous).
static __device__ inline v16h load_B_f16(const _Float16* base, int stride, int lane, int k0) {
  const int col = lane & 15;
  const int hi  = lane >> 4;
  const _Float16* p = base + (size_t)col * stride + k0 + hi * 16;
  return cat8(*(const v8h*)p, *(const v8h*)(p + 8));
}

// ---------------- kernels ----------------

__global__ void f32_to_f16_kernel(const float* __restrict__ src,
                                  _Float16* __restrict__ dst, int n) {
  for (int i = blockIdx.x * blockDim.x + threadIdx.x; i < n; i += gridDim.x * blockDim.x)
    dst[i] = (_Float16)src[i];
}

// zero h ping-pong buffers (rows 8..15 must stay zero; h0 = 0) + reset grid barrier
__global__ void init_kernel(unsigned* __restrict__ hbuf_u32, unsigned* __restrict__ bar) {
  int i = blockIdx.x * blockDim.x + threadIdx.x;
  if (i < (2 * 16 * HID * 2) / 4) hbuf_u32[i] = 0u;
  if (i == 0) { bar[0] = 0u; bar[1] = 0u; }
}

// xp[t][b][h] = x[b][t][:] . W_ih[h][:] + b_ih[h] + b_hh[h]
// One wave per 16x16 tile. M rows flatten (b,t): r = b*T + t.
__global__ __launch_bounds__(256) void xp_gemm_kernel(
    const _Float16* __restrict__ xf, const _Float16* __restrict__ wih,
    const float* __restrict__ b_ih, const float* __restrict__ b_hh,
    float* __restrict__ xp) {
  const int wave = blockIdx.x * (blockDim.x >> 5) + (threadIdx.x >> 5);
  const int lane = threadIdx.x & 31;
  const int mt = wave >> 7;          // 0..255
  const int nt = wave & 127;         // 0..127
  const int r0 = mt * 16, n0 = nt * 16;

  v8f acc0 = {}, acc1 = {};
  for (int k0 = 0; k0 < IN_DIM; k0 += 64) {
    v16h a0 = load_A_f16(xf + (size_t)r0 * IN_DIM, IN_DIM, lane, k0);
    v16h b0 = load_B_f16(wih + (size_t)n0 * IN_DIM, IN_DIM, lane, k0);
    acc0 = __builtin_amdgcn_wmma_f32_16x16x32_f16(false, a0, false, b0, (short)0, acc0, false, false);
    v16h a1 = load_A_f16(xf + (size_t)r0 * IN_DIM, IN_DIM, lane, k0 + 32);
    v16h b1 = load_B_f16(wih + (size_t)n0 * IN_DIM, IN_DIM, lane, k0 + 32);
    acc1 = __builtin_amdgcn_wmma_f32_16x16x32_f16(false, a1, false, b1, (short)0, acc1, false, false);
  }
  const int col = lane & 15, hi = lane >> 4;
  const int N = n0 + col;
  const float bias = b_ih[N] + b_hh[N];
#pragma unroll
  for (int i = 0; i < 8; ++i) {
    const int r = r0 + i + 8 * hi;
    const int b = r >> 9;            // / TSTEPS
    const int t = r & (TSTEPS - 1);
    xp[((size_t)t * BATCH + b) * HID + N] = acc0[i] + acc1[i] + bias;
  }
}

__device__ inline void grid_barrier(unsigned* cnt, unsigned* gen, unsigned nwg) {
  __syncthreads();
  if (threadIdx.x == 0) {
    const unsigned g = __hip_atomic_load(gen, __ATOMIC_RELAXED, __HIP_MEMORY_SCOPE_AGENT);
    const unsigned a = __hip_atomic_fetch_add(cnt, 1u, __ATOMIC_ACQ_REL, __HIP_MEMORY_SCOPE_AGENT);
    if (a == nwg - 1u) {
      __hip_atomic_store(cnt, 0u, __ATOMIC_RELAXED, __HIP_MEMORY_SCOPE_AGENT);
      __hip_atomic_fetch_add(gen, 1u, __ATOMIC_RELEASE, __HIP_MEMORY_SCOPE_AGENT);
    } else {
      while (__hip_atomic_load(gen, __ATOMIC_ACQUIRE, __HIP_MEMORY_SCOPE_AGENT) == g)
        __builtin_amdgcn_s_sleep(2);
    }
  }
  __syncthreads();
}

// Persistent recurrence: 32 WGs x 128 threads (4 waves). WG g owns h columns
// [g*64, g*64+64); its W_hh slice (64 rows x 2048 K, f16) is DMAed into LDS
// once by the Tensor Data Mover and stays resident for all 512 steps.
// h state (16x2048 f16, rows 8..15 zero) ping-pongs through L2.
#define NSLICE   64
#define LDS_KPAD (HID + 8)     // TDM pad: +1 DWORD per 256 DWORDs -> row stride 4112 B
__global__ __launch_bounds__(128) void rnn_persistent_kernel(
    const _Float16* __restrict__ whh, const float* __restrict__ xp,
    _Float16* __restrict__ hbuf, _Float16* __restrict__ hs,
    unsigned* __restrict__ bar) {
  __shared__ __attribute__((aligned(128))) _Float16 lds_w[NSLICE * LDS_KPAD]; // 263,168 B

  const int tid  = threadIdx.x;
  const int lane = tid & 31;
  const int w    = tid >> 5;                 // wave 0..3
  const int n_base = blockIdx.x * NSLICE;
  const int n0 = n_base + w * 16;
  const int col = lane & 15, hi = lane >> 4;

  // --- One-shot TDM DMA: 64 rows x 2048 halves of W_hh -> LDS, with hardware
  // --- pad of 1 DWORD per 256 DWORDs (bank stagger). Wave 0 issues; TDM
  // --- ignores EXEC and issues once per wave.
  if (w == 0) {
    const unsigned long long ga = (unsigned long long)(uintptr_t)(whh + (size_t)n_base * HID);
    const unsigned lds_base = (unsigned)(uintptr_t)(&lds_w[0]);
    v4u g0;
    g0.x = 1u;                                          // count=1 (valid), user mode
    g0.y = lds_base;                                    // lds_addr
    g0.z = (unsigned)(ga & 0xFFFFFFFFu);                // global_addr[31:0]
    g0.w = (unsigned)((ga >> 32) & 0x1FFFFFFu) | (2u << 30); // global_addr[56:32] | type=2
    v8i g1;
    g1[0] = (1 << 16) | (1 << 20) | (7 << 22);          // data_size=2B, pad_enable, pad_interval=256DW, pad_amount=1DW
    g1[1] = (int)(((unsigned)HID & 0xFFFFu) << 16);     // tensor_dim0 = 2048 (low16)
    g1[2] = (int)((unsigned)NSLICE << 16);              // tensor_dim0 hi=0 | tensor_dim1 = 64 (low16)
    g1[3] = (int)(((unsigned)HID & 0xFFFFu) << 16);     // tensor_dim1 hi=0 | tile_dim0 = 2048
    g1[4] = NSLICE;                                     // tile_dim1 = 64, tile_dim2 = 0
    g1[5] = HID;                                        // tensor_dim0_stride = 2048 (low32)
    g1[6] = 0;                                          // stride hi | tensor_dim1_stride lo
    g1[7] = 0;
    v4i gz = {0, 0, 0, 0};
#if __clang_major__ >= 23
    v8i gz8 = {0, 0, 0, 0, 0, 0, 0, 0};
    __builtin_amdgcn_tensor_load_to_lds(g0, g1, gz, gz, gz8, 0);
#else
    __builtin_amdgcn_tensor_load_to_lds(g0, g1, gz, gz, 0);
#endif
    __builtin_amdgcn_s_wait_tensorcnt(0);
  }
  __syncthreads();

  for (int t = 0; t < TSTEPS; ++t) {
    const _Float16* cur = hbuf + (size_t)(t & 1) * 16 * HID;
    _Float16*       nxt = hbuf + (size_t)((t + 1) & 1) * 16 * HID;

    v8f acc0 = {}, acc1 = {};
#pragma unroll 2
    for (int ks = 0; ks < HID / 64; ++ks) {
      const int k0 = ks * 64;
      v16h a0 = load_A_f16(cur, HID, lane, k0);                 // h_prev from L2
      // TDM pad skew: half c of a row lives at LDS offset c + 2*(c>>9)
      const int c0 = k0 + hi * 16;
      const _Float16* bp0 = &lds_w[(w * 16 + col) * LDS_KPAD + c0 + 2 * (c0 >> 9)];
      v16h b0 = cat8(*(const v8h*)bp0, *(const v8h*)(bp0 + 8)); // W_hh from LDS
      acc0 = __builtin_amdgcn_wmma_f32_16x16x32_f16(false, a0, false, b0, (short)0, acc0, false, false);
      v16h a1 = load_A_f16(cur, HID, lane, k0 + 32);
      const int c1 = c0 + 32;
      const _Float16* bp1 = &lds_w[(w * 16 + col) * LDS_KPAD + c1 + 2 * (c1 >> 9)];
      v16h b1 = cat8(*(const v8h*)bp1, *(const v8h*)(bp1 + 8));
      acc1 = __builtin_amdgcn_wmma_f32_16x16x32_f16(false, a1, false, b1, (short)0, acc1, false, false);
    }

    if (hi == 0) {  // lanes 0..15 hold the real batch rows M=0..7
      const int N = n0 + col;
#pragma unroll
      for (int i = 0; i < BATCH; ++i) {
        const float v = tanhf(acc0[i] + acc1[i] + xp[((size_t)t * BATCH + i) * HID + N]);
        const _Float16 hv = (_Float16)v;
        nxt[(size_t)i * HID + N] = hv;
        hs[((size_t)i * TSTEPS + t) * HID + N] = hv;
      }
      if (t + 1 < TSTEPS)
        __builtin_prefetch(&xp[((size_t)(t + 1) * BATCH) * HID + N], 0, 0);
    }
    grid_barrier(bar, bar + 1, gridDim.x);
  }
}

// out[r][o] = hs[r][:] . W_out[o][:] + b_out[o], r = b*T + t
__global__ __launch_bounds__(256) void out_gemm_kernel(
    const _Float16* __restrict__ hsf, const _Float16* __restrict__ wout,
    const float* __restrict__ b_out, float* __restrict__ out) {
  const int wave = blockIdx.x * (blockDim.x >> 5) + (threadIdx.x >> 5);
  const int lane = threadIdx.x & 31;
  const int mt = wave >> 3;          // 0..255
  const int nt = wave & 7;           // 0..7
  const int r0 = mt * 16, n0 = nt * 16;

  v8f acc0 = {}, acc1 = {};
  for (int k0 = 0; k0 < HID; k0 += 64) {
    v16h a0 = load_A_f16(hsf + (size_t)r0 * HID, HID, lane, k0);
    v16h b0 = load_B_f16(wout + (size_t)n0 * HID, HID, lane, k0);
    acc0 = __builtin_amdgcn_wmma_f32_16x16x32_f16(false, a0, false, b0, (short)0, acc0, false, false);
    v16h a1 = load_A_f16(hsf + (size_t)r0 * HID, HID, lane, k0 + 32);
    v16h b1 = load_B_f16(wout + (size_t)n0 * HID, HID, lane, k0 + 32);
    acc1 = __builtin_amdgcn_wmma_f32_16x16x32_f16(false, a1, false, b1, (short)0, acc1, false, false);
  }
  const int col = lane & 15, hi = lane >> 4;
  const int N = n0 + col;
  const float bias = b_out[N];
#pragma unroll
  for (int i = 0; i < 8; ++i)
    out[(size_t)(r0 + i + 8 * hi) * OUT_DIM + N] = acc0[i] + acc1[i] + bias;
}

// ---------------- host ----------------

extern "C" void kernel_launch(void* const* d_in, const int* in_sizes, int n_in,
                              void* d_out, int out_size, void* d_ws, size_t ws_size,
                              hipStream_t stream) {
  (void)in_sizes; (void)n_in; (void)out_size; (void)ws_size;
  const float* x     = (const float*)d_in[0];
  const float* W_ih  = (const float*)d_in[1];
  const float* b_ih  = (const float*)d_in[2];
  const float* W_hh  = (const float*)d_in[3];
  const float* b_hh  = (const float*)d_in[4];
  const float* W_out = (const float*)d_in[5];
  const float* b_out = (const float*)d_in[6];
  float* out = (float*)d_out;

  // workspace carve-up (256 B aligned)
  char* ws = (char*)d_ws;
  size_t off = 0;
  auto carve = [&](size_t bytes) { char* p = ws + off; off = (off + bytes + 255) & ~(size_t)255; return p; };
  _Float16* whh_f16  = (_Float16*)carve((size_t)HID * HID * 2);        // 8 MB
  _Float16* x_f16    = (_Float16*)carve((size_t)MROWS * IN_DIM * 2);   // 4 MB
  _Float16* wih_f16  = (_Float16*)carve((size_t)HID * IN_DIM * 2);     // 2 MB
  _Float16* wout_f16 = (_Float16*)carve((size_t)OUT_DIM * HID * 2);    // 0.5 MB
  float*    xp       = (float*)carve((size_t)TSTEPS * BATCH * HID * 4);// 32 MB
  _Float16* hbuf     = (_Float16*)carve((size_t)2 * 16 * HID * 2);     // 128 KB
  _Float16* hs_f16   = (_Float16*)carve((size_t)MROWS * HID * 2);      // 16 MB
  unsigned* bar      = (unsigned*)carve(256);

  // 1) f32 -> f16 conversions
  f32_to_f16_kernel<<<1024, 256, 0, stream>>>(x,     x_f16,    MROWS * IN_DIM);
  f32_to_f16_kernel<<<1024, 256, 0, stream>>>(W_ih,  wih_f16,  HID * IN_DIM);
  f32_to_f16_kernel<<<1024, 256, 0, stream>>>(W_hh,  whh_f16,  HID * HID);
  f32_to_f16_kernel<<<1024, 256, 0, stream>>>(W_out, wout_f16, OUT_DIM * HID);

  // 2) zero h state + barrier
  init_kernel<<<128, 256, 0, stream>>>((unsigned*)hbuf, bar);

  // 3) input projection: 256x128 tiles, 8 waves/block
  xp_gemm_kernel<<<(256 * 128) / 8, 256, 0, stream>>>(x_f16, wih_f16, b_ih, b_hh, xp);

  // 4) persistent recurrence: 32 WGs pin W_hh in LDS via TDM, grid-sync per step
  rnn_persistent_kernel<<<HID / NSLICE, 128, 0, stream>>>(whh_f16, xp, hbuf, hs_f16, bar);

  // 5) output projection: 256x8 tiles, 8 waves/block
  out_gemm_kernel<<<(256 * 8) / 8, 256, 0, stream>>>(hs_f16, wout_f16, b_out, out);
}
